// IDMForwardSim_30030411333927
// MI455X (gfx1250) — matrix operand order repaired and checked by reading, not compiled, not run on hardware
//
#include <hip/hip_runtime.h>

typedef __attribute__((ext_vector_type(16))) _Float16 v16h;
typedef __attribute__((ext_vector_type(8)))  float    v8f;
typedef __attribute__((ext_vector_type(8)))  unsigned v8u;

#define T_STEPS   50
#define BATCH     8192
#define NDEC      128
#define NIN       267
#define ROW_S     (51*14)
#define ROW_MC    (50*3)
#define O_DISP    0
#define O_ACT     (BATCH*51)
#define O_F       (O_ACT + BATCH*50)
#define O_M       (O_F + BATCH*50)
#define NFRAG     44
#define NTHREADS  32
// LDS fragment slots: [0..31] W2^T (mt*4+kc), [32..39] W1env^T (mt), [40..43] Wsc^T (kc)

static __device__ __forceinline__ v8f wmma16(v16h a, v16h b, v8f c) {
  return __builtin_amdgcn_wmma_f32_16x16x32_f16(false, a, false, b, (short)0, c, false, false);
}
static __device__ __forceinline__ unsigned pk2(float lo, float hi) {
  auto p = __builtin_amdgcn_cvt_pkrtz(lo, hi);   // __fp16 ext_vector(2), 4 bytes
  return __builtin_bit_cast(unsigned, p);
}
static __device__ __forceinline__ float leaky(float x) { return fmaxf(x, 0.3f*x); }
static __device__ __forceinline__ float clampf(float x, float lo, float hi) {
  return fminf(fmaxf(x, lo), hi);
}

struct IdmP { float tgap, jamx, maxa, inv_gden, inv_dv; };
static __device__ __forceinline__ float idm_act(float vel, float dv, float dx, const IdmP& p) {
  float dxc = clampf(dx, 0.1f, 500.f);
  float gap = p.tgap*vel + vel*dv*p.inv_gden;
  float dg  = p.jamx + fmaxf(gap, 0.f);
  float vr  = vel * p.inv_dv;
  float vr2 = vr*vr;
  float q   = dg / dxc;
  return clampf(p.maxa * (1.f - vr2*vr2 - q*q), -6.f, 6.f);
}

// Fuse one C-tile pair (features 32c..32c+15 / ..+31 on M, batch on lanes) into
// one B-layout f16 fragment for the next layer: leaky-relu, pack f32 pairs to
// f16, and exchange half the data between lane l and l+16.
static __device__ __forceinline__ v16h fuse_pair(v8f a0, v8f a1, bool lo16) {
  unsigned p0[4], p1[4];
#pragma unroll
  for (int r = 0; r < 4; ++r) {
    p0[r] = pk2(leaky(a0[2*r]), leaky(a0[2*r+1]));
    p1[r] = pk2(leaky(a1[2*r]), leaky(a1[2*r+1]));
  }
  unsigned lo[4], hi[4];
#pragma unroll
  for (int r = 0; r < 4; ++r) {
    unsigned x0 = (unsigned)__shfl_xor((int)p0[r], 16, 32);
    unsigned x1 = (unsigned)__shfl_xor((int)p1[r], 16, 32);
    lo[r] = lo16 ? p0[r] : x1;
    hi[r] = lo16 ? x0    : p1[r];
  }
  v8u u = {lo[0],lo[1],lo[2],lo[3],hi[0],hi[1],hi[2],hi[3]};
  return __builtin_bit_cast(v16h, u);
}

__global__ __launch_bounds__(NTHREADS)
void IDMForwardSim_30030411333927_kernel(
    const float* __restrict__ idm_params, const float* __restrict__ proj,
    const float* __restrict__ ench, const float* __restrict__ idm_s,
    const float* __restrict__ mcs, const float* __restrict__ smean_g,
    const float* __restrict__ svar_g, const float* __restrict__ W1,
    const float* __restrict__ b1, const float* __restrict__ W2,
    const float* __restrict__ b2, const float* __restrict__ Wf,
    const float* __restrict__ bf, const float* __restrict__ Wm,
    const float* __restrict__ bm, float* __restrict__ out)
{
  __shared__ __align__(32) _Float16 fragLDS[NFRAG*512];  // 44 KB, private to this 1-wave block
  const int tid  = threadIdx.x;
  const int lane = tid & 31;
  const bool lo16 = (lane < 16);
  const int row  = blockIdx.x*16 + (lane & 15);   // one 16-row batch tile per wave/block
  const v16h* fragV = (const v16h*)fragLDS;

  // ---- per-lane constants ----
  float sm[8], si[8];
#pragma unroll
  for (int j = 0; j < 8; ++j) { sm[j] = smean_g[j]; si[j] = 1.f/sqrtf(svar_g[j]); }
  IdmP P;
  {
    const float* pp = idm_params + row*5;
    float dv0 = pp[0]; P.tgap = pp[1]; P.jamx = pp[2]; P.maxa = pp[3];
    float mina = pp[4];
    P.inv_gden = 1.f/(2.f*sqrtf(P.maxa*mina));
    P.inv_dv = 1.f/dv0;
  }
  const float bf0 = bf[0], bm0 = bm[0];
  const int off8 = lo16 ? 0 : 8;

  // bias tiles in C/D layout (kept in registers for the whole rollout)
  v8f baseacc[8], bias2[8];
#pragma unroll
  for (int mt = 0; mt < 8; ++mt)
#pragma unroll
    for (int r = 0; r < 8; ++r) {
      baseacc[mt][r] = b1[mt*16 + off8 + r];
      bias2[mt][r]   = b2[mt*16 + off8 + r];
    }

  // ---- batch-constant input B-fragments: [proj; enc]^T, 8 K-chunks ----
  v16h bchunk[8];
  {
    const int kof = lo16 ? 0 : 16;
#pragma unroll
    for (int kc = 0; kc < 8; ++kc) {
      v16h v;
#pragma unroll
      for (int j = 0; j < 16; ++j) {
        int i = kc*32 + kof + j;
        float x = (i < 128) ? proj[row*128 + i] : ench[row*128 + (i-128)];
        v[j] = (_Float16)x;
      }
      bchunk[kc] = v;
    }
  }

  // ---- one-time: base = W1^T[:,0:256] @ [proj;enc]^T + b1 (staged, 2 rounds) ----
#pragma unroll
  for (int rnd = 0; rnd < 2; ++rnd) {
    for (int e = tid; e < 32*512; e += NTHREADS) {
      int slot = e >> 9, w = e & 511;
      int ln = w >> 4, hf = w & 15;
      int mt = slot >> 2, kc = rnd*4 + (slot & 3);
      int m = ln & 15;
      int kk = hf + ((ln >= 16) ? 16 : 0);
      fragLDS[e] = (_Float16)W1[(kc*32 + kk)*NDEC + mt*16 + m];
    }
    __syncthreads();
#pragma unroll
    for (int kcl = 0; kcl < 4; ++kcl)
#pragma unroll
      for (int mt = 0; mt < 8; ++mt)
        baseacc[mt] = wmma16(fragV[(mt*4+kcl)*32 + lane], bchunk[rnd*4+kcl], baseacc[mt]);
    __syncthreads();
  }

  // ---- stage persistent A-fragments: W2^T, W1env^T (rows 256..266), [Wf|Wm]^T ----
  for (int e = tid; e < NFRAG*512; e += NTHREADS) {
    int slot = e >> 9, w = e & 511;
    int ln = w >> 4, hf = w & 15;
    int m = ln & 15;
    int kk = hf + ((ln >= 16) ? 16 : 0);
    float val;
    if (slot < 32) {
      int mt = slot >> 2, kc = slot & 3;
      val = W2[(kc*32 + kk)*NDEC + mt*16 + m];
    } else if (slot < 40) {
      int mt = slot - 32, i = 256 + kk;
      val = (i < NIN) ? W1[i*NDEC + mt*16 + m] : 0.f;
    } else {
      int i = (slot - 40)*32 + kk;
      val = (m == 0) ? Wf[i] : ((m == 1) ? Wm[i] : 0.f);
    }
    fragLDS[e] = (_Float16)val;
  }
  __syncthreads();

  // ---- init rollout state (lanes 16..31 mirror lane-16's row; outputs gated) ----
  const float* s0 = idm_s + row*ROW_S;
  float ego_v = s0[0], ego_a = s0[11], ego_x = s0[3], disp = 0.f;
  if (lo16) out[O_DISP + row*51] = 0.f;

  v8u zu = {0,0,0,0,0,0,0,0};
  const v16h hz = __builtin_bit_cast(v16h, zu);

  for (int t = 0; t < T_STEPS; ++t) {
    const float* sp = idm_s + row*ROW_S + t*14;
    float f_v = sp[1], m_v = sp[2], f_x = sp[4], m_x = sp[5];
    float f_a = sp[12], m_e = sp[13];
    const float* mp = mcs + row*ROW_MC + t*3;
    float mc0 = mp[0], mc1 = mp[1], mc2 = mp[2];

    float ef_dx = f_x - ego_x;
    float em_dx = (m_x - ego_x)*m_e + (1.f - m_e)*100.f;
    float ef_dv = ego_v - f_v;
    float em_dv = (ego_v - m_v)*m_e;

    float ev[8] = {ego_a, f_a, ego_v, f_v, ef_dv, ef_dx, em_dv, em_dx};
    v16h benv = hz;
#pragma unroll
    for (int j = 0; j < 8; ++j) benv[j] = (_Float16)((ev[j]-sm[j])*si[j]);
    benv[8] = (_Float16)mc0; benv[9] = (_Float16)mc1; benv[10] = (_Float16)mc2;
    benv = lo16 ? benv : hz;   // K=16..31 of the env chunk is zero padding

    // layer 1 (pairwise): h1^T = leaky(base + W1env^T @ env), fused into B-frags
    v16h bh1[4];
#pragma unroll
    for (int c = 0; c < 4; ++c) {
      v8f a0 = wmma16(fragV[(32+2*c  )*32 + lane], benv, baseacc[2*c  ]);
      v8f a1 = wmma16(fragV[(32+2*c+1)*32 + lane], benv, baseacc[2*c+1]);
      bh1[c] = fuse_pair(a0, a1, lo16);
    }

    // layer 2 (pairwise): h2^T = leaky(W2^T @ h1^T + b2), fused into B-frags
    v16h bh2[4];
#pragma unroll
    for (int c = 0; c < 4; ++c) {
      v8f a0 = bias2[2*c], a1 = bias2[2*c+1];
#pragma unroll
      for (int kc = 0; kc < 4; ++kc) {
        a0 = wmma16(fragV[((2*c  )*4+kc)*32 + lane], bh1[kc], a0);
        a1 = wmma16(fragV[((2*c+1)*4+kc)*32 + lane], bh1[kc], a1);
      }
      bh2[c] = fuse_pair(a0, a1, lo16);
    }

    // scores: rows 0/1 of [Wf|Wm]^T @ h2^T land on lane = batch row
    v8f sc = {0.f,0.f,0.f,0.f,0.f,0.f,0.f,0.f};
#pragma unroll
    for (int kc = 0; kc < 4; ++kc)
      sc = wmma16(fragV[(40+kc)*32 + lane], bh2[kc], sc);

    float f_sc = clampf(sc[0] + bf0, -20.f, 20.f);
    float m_sc = clampf(sc[1] + bm0, -20.f, 20.f);
    float f_att = 1.f/(1.f + __expf((m_sc - f_sc)*5.f));  // stable softmax pair
    float m_att = 1.f - f_att;

    float ef_act = idm_act(ego_v, ef_dv, ef_dx, P);
    float em_act = idm_act(ego_v, em_dv, em_dx, P);
    float a2 = f_att*ef_act + m_att*em_act;

    float delta = ego_v*0.1f + 0.5f*a2*0.01f;
    disp += delta; ego_x += delta; ego_v += a2*0.1f; ego_a = a2;

    if (lo16) {
      out[O_DISP + row*51 + t + 1] = disp;
      out[O_ACT  + row*50 + t]     = a2;
      out[O_F    + row*50 + t]     = f_att;
      out[O_M    + row*50 + t]     = m_att;
    }
  }
}

extern "C" void kernel_launch(void* const* d_in, const int* in_sizes, int n_in,
                              void* d_out, int out_size, void* d_ws, size_t ws_size,
                              hipStream_t stream) {
  (void)in_sizes; (void)n_in; (void)out_size; (void)d_ws; (void)ws_size;
  dim3 grid(BATCH/16), block(NTHREADS);   // 512 one-wave blocks -> spread over all WGPs
  IDMForwardSim_30030411333927_kernel<<<grid, block, 0, stream>>>(
      (const float*)d_in[0],  (const float*)d_in[1],  (const float*)d_in[2],
      (const float*)d_in[3],  (const float*)d_in[4],  (const float*)d_in[5],
      (const float*)d_in[6],  (const float*)d_in[7],  (const float*)d_in[8],
      (const float*)d_in[9],  (const float*)d_in[10], (const float*)d_in[11],
      (const float*)d_in[12], (const float*)d_in[13], (const float*)d_in[14],
      (float*)d_out);
}
